// SHSA_87196426043780
// MI455X (gfx1250) — matrix operand order repaired and checked by reading, not compiled, not run on hardware
//
#include <hip/hip_runtime.h>
#include <hip/hip_bf16.h>
#include <stdint.h>

#define DIMC 384
#define PDC  96
#define QKC  16
#define VDC  96
#define QKVO 128
#define NPOS 4096
#define EPSF 1e-5f

typedef __attribute__((ext_vector_type(16))) __bf16 v16bf;
typedef __attribute__((ext_vector_type(8)))  float  v8f;

union BFrag { uint4 q[2]; v16bf v; };

__device__ __forceinline__ __bf16 f2bf(float f) {
    union { float f; uint32_t u; } a; a.f = f;
    uint32_t r = a.u + 0x7FFFu + ((a.u >> 16) & 1u);
    uint16_t h = (uint16_t)(r >> 16);
    return __builtin_bit_cast(__bf16, h);
}

__device__ __forceinline__ v8f vzero8() {
    v8f z;
#pragma unroll
    for (int i = 0; i < 8; ++i) z[i] = 0.0f;
    return z;
}

// ---------------------------------------------------------------------------
// Kernel 0: fold BN params, repack weights into WMMA A-fragment order (bf16).
// A-frag element map: lane L holds row M = L%16; vector elem e holds
// K = (L/16)*8 + (e%8) + (e>=8 ? 16 : 0) within a 32-wide K chunk.
// ---------------------------------------------------------------------------
__global__ void shsa_prep(const float* __restrict__ w_qkv,
                          const float* __restrict__ g1, const float* __restrict__ b1,
                          const float* __restrict__ m1, const float* __restrict__ v1,
                          const float* __restrict__ w_proj,
                          const float* __restrict__ g2, const float* __restrict__ b2,
                          const float* __restrict__ m2, const float* __restrict__ v2,
                          float* __restrict__ inv1, float* __restrict__ beta1,
                          float* __restrict__ inv2, float* __restrict__ beta2,
                          __bf16* __restrict__ wq1, __bf16* __restrict__ wq3) {
    const int tid = threadIdx.x;
    if (tid < QKVO) {
        float iv = g1[tid] * rsqrtf(v1[tid] + EPSF);
        inv1[tid] = iv; beta1[tid] = b1[tid] - m1[tid] * iv;
    }
    if (tid < DIMC) {
        float iv = g2[tid] * rsqrtf(v2[tid] + EPSF);
        inv2[tid] = iv; beta2[tid] = b2[tid] - m2[tid] * iv;
    }
    // wq1: [mt=8][kc=3][lane=32][e=16]
    for (int idx = tid; idx < 8 * 3 * 512; idx += 256) {
        int e = idx & 15, lane = (idx >> 4) & 31, kc = (idx >> 9) % 3, mt = idx / 1536;
        int c  = kc * 32 + (lane >> 4) * 8 + (e & 7) + ((e >= 8) ? 16 : 0);
        int oc = mt * 16 + (lane & 15);
        wq1[idx] = f2bf(w_qkv[oc * PDC + c]);
    }
    // wq3: [mt=24][kc=12][lane=32][e=16]
    for (int idx = tid; idx < 24 * 12 * 512; idx += 256) {
        int e = idx & 15, lane = (idx >> 4) & 31, kc = (idx >> 9) % 12, mt = idx / 6144;
        int c  = kc * 32 + (lane >> 4) * 8 + (e & 7) + ((e >= 8) ? 16 : 0);
        int oc = mt * 16 + (lane & 15);
        wq3[idx] = f2bf(w_proj[oc * DIMC + c]);
    }
}

// ---------------------------------------------------------------------------
// Kernel 1: LayerNorm + QKV GEMM (WMMA) + BN1. One block = 64 positions.
// Outputs: qbuf[b][n][16] (scale folded), kbuf[b][n][16], vt[b][vd][n]  (bf16)
// ---------------------------------------------------------------------------
__global__ void shsa_lnqkv(const float* __restrict__ x,
                           const float* __restrict__ ln_w, const float* __restrict__ ln_b,
                           const float* __restrict__ inv1, const float* __restrict__ beta1,
                           const __bf16* __restrict__ wq1,
                           __bf16* __restrict__ qbuf, __bf16* __restrict__ kbuf,
                           __bf16* __restrict__ vt) {
    __shared__ __align__(16) float  xs[PDC][65];
    __shared__ float mu[64], rstd[64];
    __shared__ __align__(16) __bf16 xnt[64][104];   // [n][c], padded
    const int tid = threadIdx.x;
    const int b  = blockIdx.x >> 6;
    const int n0 = (blockIdx.x & 63) << 6;
    const size_t xbase = ((size_t)b * DIMC) * NPOS + n0;

    for (int idx = tid; idx < PDC * 64; idx += 256) {
        int c = idx >> 6, n = idx & 63;
        xs[c][n] = x[xbase + (size_t)c * NPOS + n];
    }
    __syncthreads();
    if (tid < 64) {
        float s = 0.f, s2 = 0.f;
        for (int c = 0; c < PDC; ++c) { float v = xs[c][tid]; s += v; s2 += v * v; }
        float m = s * (1.0f / PDC);
        float var = s2 * (1.0f / PDC) - m * m;
        mu[tid] = m; rstd[tid] = rsqrtf(var + EPSF);
    }
    __syncthreads();
    for (int idx = tid; idx < PDC * 64; idx += 256) {
        int c = idx >> 6, n = idx & 63;
        xnt[n][c] = f2bf((xs[c][n] - mu[n]) * rstd[n] * ln_w[c] + ln_b[c]);
    }
    __syncthreads();

    const int wav = tid >> 5, lane = tid & 31;
    const int half = lane >> 4, l15 = lane & 15;
    const int mt = wav;                         // 8 waves cover 128 out-channels
    v8f acc[4];
#pragma unroll
    for (int i = 0; i < 4; ++i) acc[i] = vzero8();
    for (int kc = 0; kc < 3; ++kc) {
        BFrag a;
        const uint4* ap = (const uint4*)(wq1 + ((size_t)((mt * 3 + kc) * 32 + lane)) * 16);
        a.q[0] = ap[0]; a.q[1] = ap[1];
#pragma unroll
        for (int nt = 0; nt < 4; ++nt) {
            BFrag bb;
            const uint4* bp = (const uint4*)&xnt[nt * 16 + l15][kc * 32 + half * 8];
            bb.q[0] = bp[0]; bb.q[1] = bp[2];
            acc[nt] = __builtin_amdgcn_wmma_f32_16x16x32_bf16(
                false, a.v, false, bb.v, (short)0, acc[nt], false, false);
        }
    }
#pragma unroll
    for (int nt = 0; nt < 4; ++nt) {
        int n = n0 + nt * 16 + l15;
#pragma unroll
        for (int r = 0; r < 8; ++r) {
            int oc = mt * 16 + r + half * 8;
            float val = acc[nt][r] * inv1[oc] + beta1[oc];
            if (oc < QKC) {
                qbuf[((size_t)b * NPOS + n) * QKC + oc] = f2bf(val * 0.25f); // fold 1/sqrt(16)
            } else if (oc < 2 * QKC) {
                kbuf[((size_t)b * NPOS + n) * QKC + (oc - QKC)] = f2bf(val);
            } else {
                vt[((size_t)b * VDC + (oc - 2 * QKC)) * NPOS + n] = f2bf(val);
            }
        }
    }
}

// ---------------------------------------------------------------------------
// Kernel 2: flash attention. Block = 4 waves, each wave owns 16 queries.
// Per 32-key chunk: 2 QK^T WMMAs (K padded 16->32), online softmax in D-layout,
// P through LDS -> A-fragment, 6 PV WMMAs. ReLU on writeout to xmo[b][vd][n].
// ---------------------------------------------------------------------------
__global__ void shsa_attn(const __bf16* __restrict__ qbuf, const __bf16* __restrict__ kbuf,
                          const __bf16* __restrict__ vt, __bf16* __restrict__ xmo) {
    __shared__ __align__(16) __bf16 vs[VDC][32];   // [vd][key]
    __shared__ __align__(16) float  ps[4][16][32]; // per-wave P scratch [q][key]
    const int tid  = threadIdx.x;
    const int lane = tid & 31, wav = tid >> 5;
    const int half = lane >> 4, l15 = lane & 15;
    const int b = blockIdx.x >> 6;
    const int qbase = ((blockIdx.x & 63) << 6) + wav * 16;

    BFrag qa; // Q A-fragment, K-dim 16 padded to 32 with zeros
    {
        const uint4* qp = (const uint4*)(qbuf + ((size_t)b * NPOS + qbase + l15) * QKC);
        qa.q[0] = half ? make_uint4(0, 0, 0, 0) : qp[0];
        qa.q[1] = half ? make_uint4(0, 0, 0, 0) : qp[1];
    }
    float mrun[8], lrun[8];
    v8f acc[6];
#pragma unroll
    for (int r = 0; r < 8; ++r) { mrun[r] = -3.0e38f; lrun[r] = 0.f; }
#pragma unroll
    for (int i = 0; i < 6; ++i) acc[i] = vzero8();

    for (int kk = 0; kk < NPOS; kk += 32) {
        __syncthreads();
        for (int idx = tid; idx < VDC * 4; idx += 128) {      // stage V chunk
            int row = idx >> 2, seg = idx & 3;
            const uint4* sp = (const uint4*)(vt + ((size_t)b * VDC + row) * NPOS + kk + seg * 8);
            *(uint4*)&vs[row][seg * 8] = sp[0];
        }
        __syncthreads();

        v8f s0, s1;
        {
            BFrag kf;
            if (lane < 16) {
                const uint4* kp = (const uint4*)(kbuf + ((size_t)b * NPOS + kk + l15) * QKC);
                kf.q[0] = kp[0]; kf.q[1] = kp[1];
            } else { kf.q[0] = make_uint4(0,0,0,0); kf.q[1] = make_uint4(0,0,0,0); }
            s0 = __builtin_amdgcn_wmma_f32_16x16x32_bf16(
                false, qa.v, false, kf.v, (short)0, vzero8(), false, false);
        }
        {
            BFrag kf;
            if (lane < 16) {
                const uint4* kp = (const uint4*)(kbuf + ((size_t)b * NPOS + kk + 16 + l15) * QKC);
                kf.q[0] = kp[0]; kf.q[1] = kp[1];
            } else { kf.q[0] = make_uint4(0,0,0,0); kf.q[1] = make_uint4(0,0,0,0); }
            s1 = __builtin_amdgcn_wmma_f32_16x16x32_bf16(
                false, qa.v, false, kf.v, (short)0, vzero8(), false, false);
        }

        float alpha[8];
#pragma unroll
        for (int r = 0; r < 8; ++r) {   // online softmax, rows r + half*8
            float m = fmaxf(s0[r], s1[r]);
            m = fmaxf(m, __shfl_xor(m, 1, 32));
            m = fmaxf(m, __shfl_xor(m, 2, 32));
            m = fmaxf(m, __shfl_xor(m, 4, 32));
            m = fmaxf(m, __shfl_xor(m, 8, 32));
            float mn = fmaxf(mrun[r], m);
            alpha[r] = __expf(mrun[r] - mn);
            mrun[r] = mn;
            float p0 = __expf(s0[r] - mn);
            float p1 = __expf(s1[r] - mn);
            s0[r] = p0; s1[r] = p1;
            float rsum = p0 + p1;
            rsum += __shfl_xor(rsum, 1, 32);
            rsum += __shfl_xor(rsum, 2, 32);
            rsum += __shfl_xor(rsum, 4, 32);
            rsum += __shfl_xor(rsum, 8, 32);
            lrun[r] = lrun[r] * alpha[r] + rsum;
        }
#pragma unroll
        for (int r = 0; r < 8; ++r) {   // D-layout -> LDS [q][key]
            ps[wav][r + half * 8][l15]      = s0[r];
            ps[wav][r + half * 8][16 + l15] = s1[r];
        }
        __syncthreads();

        BFrag pf;   // P as A-fragment (bf16)
        {
            const int c0 = half * 8;
            const float4 p0 = *(const float4*)&ps[wav][l15][c0];
            const float4 p1 = *(const float4*)&ps[wav][l15][c0 + 4];
            const float4 p2 = *(const float4*)&ps[wav][l15][16 + c0];
            const float4 p3 = *(const float4*)&ps[wav][l15][16 + c0 + 4];
            pf.v[0]=f2bf(p0.x); pf.v[1]=f2bf(p0.y); pf.v[2]=f2bf(p0.z); pf.v[3]=f2bf(p0.w);
            pf.v[4]=f2bf(p1.x); pf.v[5]=f2bf(p1.y); pf.v[6]=f2bf(p1.z); pf.v[7]=f2bf(p1.w);
            pf.v[8]=f2bf(p2.x); pf.v[9]=f2bf(p2.y); pf.v[10]=f2bf(p2.z); pf.v[11]=f2bf(p2.w);
            pf.v[12]=f2bf(p3.x); pf.v[13]=f2bf(p3.y); pf.v[14]=f2bf(p3.z); pf.v[15]=f2bf(p3.w);
        }
#pragma unroll
        for (int vdt = 0; vdt < 6; ++vdt) {
            BFrag vf;
            const uint4* vp = (const uint4*)&vs[vdt * 16 + l15][half * 8];
            vf.q[0] = vp[0]; vf.q[1] = vp[2];
#pragma unroll
            for (int r = 0; r < 8; ++r) acc[vdt][r] *= alpha[r];
            acc[vdt] = __builtin_amdgcn_wmma_f32_16x16x32_bf16(
                false, pf.v, false, vf.v, (short)0, acc[vdt], false, false);
        }
    }
#pragma unroll
    for (int vdt = 0; vdt < 6; ++vdt) {
        int vd = vdt * 16 + l15;
#pragma unroll
        for (int r = 0; r < 8; ++r) {
            int n = qbase + r + half * 8;
            float o = acc[vdt][r] / lrun[r];
            xmo[((size_t)b * VDC + vd) * NPOS + n] = f2bf(fmaxf(o, 0.f));
        }
    }
}

// ---------------------------------------------------------------------------
// Kernel 3: y = BN2(w_proj @ relu(concat(attn_out, x_skip))). WMMA GEMM.
// ---------------------------------------------------------------------------
__global__ void shsa_proj(const float* __restrict__ x, const __bf16* __restrict__ xmo,
                          const __bf16* __restrict__ wq3,
                          const float* __restrict__ inv2, const float* __restrict__ beta2,
                          float* __restrict__ y) {
    __shared__ __align__(16) __bf16 xmt[64][392];  // [n][c], padded (stride 784B)
    const int tid = threadIdx.x;
    const int b  = blockIdx.x >> 6;
    const int n0 = (blockIdx.x & 63) << 6;

    for (int idx = tid; idx < DIMC * 64; idx += 256) {
        int c = idx >> 6, n = idx & 63;
        __bf16 v;
        if (c < VDC) {
            v = xmo[((size_t)b * VDC + c) * NPOS + n0 + n];   // already ReLU'd
        } else {
            float f = x[((size_t)b * DIMC + c) * NPOS + n0 + n];
            v = f2bf(fmaxf(f, 0.f));
        }
        xmt[n][c] = v;
    }
    __syncthreads();

    const int wav = tid >> 5, lane = tid & 31;
    const int half = lane >> 4, l15 = lane & 15;
    for (int nt = 0; nt < 4; ++nt) {
        v8f acc[3];
#pragma unroll
        for (int i = 0; i < 3; ++i) acc[i] = vzero8();
        for (int kc = 0; kc < 12; ++kc) {
            BFrag bb;
            const uint4* bp = (const uint4*)&xmt[nt * 16 + l15][kc * 32 + half * 8];
            bb.q[0] = bp[0]; bb.q[1] = bp[2];
#pragma unroll
            for (int i = 0; i < 3; ++i) {
                int mt = wav * 3 + i;
                BFrag a;
                const uint4* ap = (const uint4*)(wq3 + ((size_t)((mt * 12 + kc) * 32 + lane)) * 16);
                a.q[0] = ap[0]; a.q[1] = ap[1];
                acc[i] = __builtin_amdgcn_wmma_f32_16x16x32_bf16(
                    false, a.v, false, bb.v, (short)0, acc[i], false, false);
            }
        }
#pragma unroll
        for (int i = 0; i < 3; ++i) {
#pragma unroll
            for (int r = 0; r < 8; ++r) {
                int oc = (wav * 3 + i) * 16 + r + half * 8;
                int n  = n0 + nt * 16 + l15;
                y[((size_t)b * DIMC + oc) * NPOS + n] = acc[i][r] * inv2[oc] + beta2[oc];
            }
        }
    }
}

// ---------------------------------------------------------------------------
extern "C" void kernel_launch(void* const* d_in, const int* in_sizes, int n_in,
                              void* d_out, int out_size, void* d_ws, size_t ws_size,
                              hipStream_t stream) {
    const float* x      = (const float*)d_in[0];
    const float* ln_w   = (const float*)d_in[1];
    const float* ln_b   = (const float*)d_in[2];
    const float* w_qkv  = (const float*)d_in[3];
    const float* bn1_g  = (const float*)d_in[4];
    const float* bn1_b  = (const float*)d_in[5];
    const float* bn1_m  = (const float*)d_in[6];
    const float* bn1_v  = (const float*)d_in[7];
    const float* w_proj = (const float*)d_in[8];
    const float* bn2_g  = (const float*)d_in[9];
    const float* bn2_b  = (const float*)d_in[10];
    const float* bn2_m  = (const float*)d_in[11];
    const float* bn2_v  = (const float*)d_in[12];
    float* y = (float*)d_out;
    (void)in_sizes; (void)n_in; (void)out_size; (void)ws_size;

    char* ws = (char*)d_ws;
    size_t off = 0;
    auto walloc = [&](size_t bytes) -> char* {
        char* p = ws + off;
        off = (off + bytes + 255) & ~(size_t)255;
        return p;
    };
    float*  inv1  = (float*) walloc(128 * 4);
    float*  beta1 = (float*) walloc(128 * 4);
    float*  inv2  = (float*) walloc(384 * 4);
    float*  beta2 = (float*) walloc(384 * 4);
    __bf16* wq1   = (__bf16*)walloc((size_t)8 * 3 * 512 * 2);
    __bf16* wq3   = (__bf16*)walloc((size_t)24 * 12 * 512 * 2);
    __bf16* qbuf  = (__bf16*)walloc((size_t)4 * NPOS * QKC * 2);
    __bf16* kbuf  = (__bf16*)walloc((size_t)4 * NPOS * QKC * 2);
    __bf16* vt    = (__bf16*)walloc((size_t)4 * VDC * NPOS * 2);
    __bf16* xmo   = (__bf16*)walloc((size_t)4 * VDC * NPOS * 2);

    hipLaunchKernelGGL(shsa_prep, dim3(1), dim3(256), 0, stream,
                       w_qkv, bn1_g, bn1_b, bn1_m, bn1_v,
                       w_proj, bn2_g, bn2_b, bn2_m, bn2_v,
                       inv1, beta1, inv2, beta2, wq1, wq3);
    hipLaunchKernelGGL(shsa_lnqkv, dim3(256), dim3(256), 0, stream,
                       x, ln_w, ln_b, inv1, beta1, wq1, qbuf, kbuf, vt);
    hipLaunchKernelGGL(shsa_attn, dim3(256), dim3(128), 0, stream,
                       qbuf, kbuf, vt, xmo);
    hipLaunchKernelGGL(shsa_proj, dim3(256), dim3(256), 0, stream,
                       x, xmo, wq3, inv2, beta2, y);
}
